// Encoder_PAC_self_67946382622797
// MI455X (gfx1250) — matrix operand order repaired and checked by reading, not compiled, not run on hardware
//
#include <hip/hip_runtime.h>
#include <hip/hip_bf16.h>

typedef __attribute__((ext_vector_type(16))) _Float16 v16h;
typedef __attribute__((ext_vector_type(8)))  float    v8f;

#define BATCH 16
#define EPS 1e-5f

// ---------------------------------------------------------------------------
// Pre-pack weights [O, C, 3, 3] into WMMA B-fragment order (f16).
// K layout is k-major: Kidx = k*CIN + c.  B frags use contiguous k (not AMD's table order):
//   lane = col N (0-15) + 16*half ; elem e -> K = half*16 + e within step.
// Fragment storage: bws[((n*STEPS + step)*32 + lane)*16 + e]
// ---------------------------------------------------------------------------
template<int CIN, int O>
__global__ void prep_b_frags(const float* __restrict__ w, _Float16* __restrict__ bws) {
  constexpr int KTOT  = CIN * 9;
  constexpr int STEPS = (KTOT + 31) / 32;
  constexpr int NT    = O / 16;
  int idx = blockIdx.x * blockDim.x + threadIdx.x;
  if (idx >= NT * STEPS * 32) return;
  int lane = idx & 31;
  int frag = idx >> 5;
  int step = frag % STEPS;
  int n    = frag / STEPS;
  int col  = lane & 15;
  int half = lane >> 4;
  int o    = n * 16 + col;
  _Float16* dst = bws + (size_t)idx * 16;
  #pragma unroll
  for (int e = 0; e < 16; ++e) {
    int Kidx = step * 32 + half * 16 + e;
    float v = 0.f;
    if (Kidx < KTOT) {
      int k = Kidx / CIN;             // patch position 0..8
      int c = Kidx % CIN;             // input channel
      v = w[((size_t)o * CIN + c) * 9 + k];
    }
    dst[e] = (_Float16)v;
  }
}

// ---------------------------------------------------------------------------
// PAC conv via WMMA, LDS-staged input window.
// Block = 128 threads (4 waves).  Block covers 64 consecutive output pixels
// of one output row; each wave owns 16 pixels x ALL O channels.
// The 3-row x (64*S+2)-col x CIN input window is staged into LDS once with
// CDNA5 async global->LDS loads (ASYNCcnt), zero-filled at pad borders; both
// the gaussian-kernel pass and the WMMA A-build pass then read LDS only.
// ---------------------------------------------------------------------------
template<int CIN, int O, int STRIDE, int H>
__global__ void __launch_bounds__(128)
pac_conv_wmma(const float* __restrict__ x, const _Float16* __restrict__ bws,
              float* __restrict__ out) {
  constexpr int KTOT  = CIN * 9;
  constexpr int STEPS = (KTOT + 31) / 32;
  constexpr int NT    = O / 16;
  constexpr int W     = H;
  constexpr int Ho    = (H - 1) / STRIDE + 1;
  constexpr int Wo    = Ho;
  constexpr int WB    = Wo / 64;               // 64-pixel blocks per row
  constexpr int WL    = 64 * STRIDE + 2;       // staged window width
  constexpr int PITCH = WL + 2;                // padded LDS row pitch

  extern __shared__ float smem[];              // [CIN][3][PITCH], LDS offset 0

  const int tid  = threadIdx.x;
  const int lane = tid & 31;
  const int wv   = tid >> 5;                   // wave id 0..3
  const int pix  = lane & 15;                  // A: row M ; B/D: col N
  const int half = lane >> 4;

  int t   = blockIdx.x;
  const int xb  = t % WB;  t /= WB;            // 64-pixel block in row
  const int row = t % Ho;
  const int b   = t / Ho;

  const int bpix = xb * 64;                    // first output pixel of block
  const int bx0  = bpix * STRIDE - 1;          // leftmost input col of window
  const int iy0  = row * STRIDE - 1;           // top input row of window
  const float* xbp = x + (size_t)b * CIN * H * W;
  const unsigned long long sbase = (unsigned long long)(uintptr_t)xbp;

  // ---- stage window into LDS: async global->LDS for in-range, zero pads ----
  {
    constexpr int NE = CIN * 3 * WL;
    for (int i = tid; i < NE; i += 128) {
      int xl = i % WL;
      int rc = i / WL;
      int ry = rc % 3;
      int c  = rc / 3;
      int gx = bx0 + xl;
      int gy = iy0 + ry;
      int li = (c * 3 + ry) * PITCH + xl;      // float index; LDS byte = li*4
      if ((unsigned)gx < (unsigned)W && (unsigned)gy < (unsigned)H) {
        unsigned loff = (unsigned)(li * 4);
        unsigned goff = (unsigned)(((c * H + gy) * W + gx) * 4);
        asm volatile("global_load_async_to_lds_b32 %0, %1, %2"
                     :: "v"(loff), "v"(goff), "s"(sbase) : "memory");
      } else {
        smem[li] = 0.f;                        // zero padding (matches jnp.pad)
      }
    }
  }
  asm volatile("s_wait_asynccnt 0x0" ::: "memory");
  __syncthreads();

  const int xl0 = (wv * 16 + pix) * STRIDE;    // this lane's patch-left in LDS

  // ---- adaptive gaussian kernel (f == x): split channels across lane halves
  float ss[9];
  #pragma unroll
  for (int k = 0; k < 9; ++k) ss[k] = 0.f;
  for (int c = half; c < CIN; c += 2) {
    const float* sc = smem + c * 3 * PITCH;
    float v[9];
    #pragma unroll
    for (int k = 0; k < 9; ++k)
      v[k] = sc[(k / 3) * PITCH + xl0 + (k % 3)];
    float ctr = v[4];
    #pragma unroll
    for (int k = 0; k < 9; ++k) { float d = v[k] - ctr; ss[k] += d * d; }
  }
  #pragma unroll
  for (int k = 0; k < 9; ++k) ss[k] += __shfl_xor(ss[k], 16, 32);
  float kern[9];
  #pragma unroll
  for (int k = 0; k < 9; ++k) kern[k] = __expf(-0.5f * ss[k]);

  // ---- K loop with v_wmma_f32_16x16x32_f16 ----
  v8f acc[NT];
  #pragma unroll
  for (int n = 0; n < NT; ++n)
    #pragma unroll
    for (int i = 0; i < 8; ++i) acc[n][i] = 0.f;

  const v16h* bfp = (const v16h*)bws;
  #pragma unroll
  for (int step = 0; step < STEPS; ++step) {
    // A layout: lane<16 elem e -> K e ; lane>=16 -> K e+16
    v16h a;
    #pragma unroll
    for (int e = 0; e < 16; ++e) {
      int kloc = half * 16 + e;
      int Kidx = step * 32 + kloc;
      float val = 0.f;
      if (Kidx < KTOT) {
        int k  = Kidx / CIN;                   // uniform per step for CIN 32/64
        int c  = Kidx % CIN;
        float xv = smem[(c * 3 + k / 3) * PITCH + xl0 + (k % 3)];
        val = xv * kern[k];
      }
      a[e] = (_Float16)val;
    }
    #pragma unroll
    for (int n = 0; n < NT; ++n) {
      v16h bf = bfp[(n * STEPS + step) * 32 + lane];
      acc[n] = __builtin_amdgcn_wmma_f32_16x16x32_f16(
          /*neg_a=*/false, a, /*neg_b=*/false, bf,
          /*c_mod=*/(short)0, acc[n], /*reuse_a=*/false, /*reuse_b=*/false);
    }
  }

  // ---- store D: vgpr elem i -> pixel (tile + 8*half + i), lane -> channel
  const int pxb = bpix + wv * 16 + half * 8;
  #pragma unroll
  for (int n = 0; n < NT; ++n) {
    int och = n * 16 + pix;
    float* op = out + (((size_t)b * O + och) * (size_t)Ho + row) * Wo + pxb;
    #pragma unroll
    for (int i = 0; i < 8; ++i) op[i] = acc[n][i];  // 8 contiguous floats
  }
}

// ---------------------------------------------------------------------------
// Fused double (inorm + residual) stats:
//   y = (x-m)*s + x,  s = rsqrt(var+eps);  mean(y)=m, var(y)=(1+s)^2*var
//   out = (y-m)*s2 + y, s2 = rsqrt((1+s)^2*var + eps)
// ---------------------------------------------------------------------------
__global__ void plane_stats(const float* __restrict__ h, float* __restrict__ stats,
                            int planeSize) {
  __shared__ float s_sum[256], s_sq[256];
  const int tid = threadIdx.x;
  const float* hp = h + (size_t)blockIdx.x * planeSize;
  float sum = 0.f, sq = 0.f;
  for (int i = tid; i < planeSize; i += 256) {
    float v = hp[i];
    sum += v; sq += v * v;
  }
  s_sum[tid] = sum; s_sq[tid] = sq;
  __syncthreads();
  for (int off = 128; off > 0; off >>= 1) {
    if (tid < off) { s_sum[tid] += s_sum[tid + off]; s_sq[tid] += s_sq[tid + off]; }
    __syncthreads();
  }
  if (tid == 0) {
    float inv = 1.f / (float)planeSize;
    float m = s_sum[0] * inv;
    float v = fmaxf(s_sq[0] * inv - m * m, 0.f);
    float s = rsqrtf(v + EPS);
    float op = 1.f + s;
    float s2 = rsqrtf(op * op * v + EPS);
    stats[blockIdx.x * 3 + 0] = m;
    stats[blockIdx.x * 3 + 1] = s;
    stats[blockIdx.x * 3 + 2] = s2;
  }
}

__global__ void inorm_apply(float* __restrict__ h, const float* __restrict__ stats,
                            int planeShift, size_t total) {
  size_t i = (size_t)blockIdx.x * blockDim.x + threadIdx.x;
  if (i >= total) return;
  int p = (int)(i >> planeShift);
  float m  = stats[p * 3 + 0];
  float s  = stats[p * 3 + 1];
  float s2 = stats[p * 3 + 2];
  float xv = h[i];
  float y  = (xv - m) * s + xv;
  h[i] = (y - m) * s2 + y;
}

// ---------------------------------------------------------------------------
extern "C" void kernel_launch(void* const* d_in, const int* in_sizes, int n_in,
                              void* d_out, int out_size, void* d_ws, size_t ws_size,
                              hipStream_t stream) {
  const float* x  = (const float*)d_in[0];
  const float* w1 = (const float*)d_in[1];
  const float* w2 = (const float*)d_in[2];
  const float* w3 = (const float*)d_in[3];

  float* out = (float*)d_out;
  float* ef1 = out;                                        // [16, 32,256,256]
  float* ef2 = ef1 + (size_t)BATCH * 32 * 256 * 256;       // [16, 64,128,128]
  float* ef3 = ef2 + (size_t)BATCH * 64 * 128 * 128;       // [16,128, 64, 64]

  _Float16* bws   = (_Float16*)d_ws;                       // <= 148 KB (stage 3)
  float*    stats = (float*)((char*)d_ws + (256 << 10));   // <= 24 KB

  // ---------------- stage 1: pac_conv(x, x, w1, stride=1) ----------------
  {
    constexpr int CIN = 3, O = 32, S = 1, H = 256, Ho = 256;
    constexpr int STEPS = (CIN * 9 + 31) / 32, NT = O / 16;
    constexpr int SMEM = CIN * 3 * (64 * S + 4) * 4;
    prep_b_frags<CIN, O><<<(NT * STEPS * 32 + 63) / 64, 64, 0, stream>>>(w1, bws);
    pac_conv_wmma<CIN, O, S, H><<<BATCH * Ho * (Ho / 64), 128, SMEM, stream>>>(x, bws, ef1);
    plane_stats<<<BATCH * O, 256, 0, stream>>>(ef1, stats, Ho * Ho);
    size_t tot = (size_t)BATCH * O * Ho * Ho;
    inorm_apply<<<(unsigned)((tot + 255) / 256), 256, 0, stream>>>(ef1, stats, 16, tot);
  }
  // ---------------- stage 2: pac_conv(ef1, ef1, w2, stride=2) ------------
  {
    constexpr int CIN = 32, O = 64, S = 2, H = 256, Ho = 128;
    constexpr int STEPS = (CIN * 9 + 31) / 32, NT = O / 16;
    constexpr int SMEM = CIN * 3 * (64 * S + 4) * 4;       // ~50.7 KB
    prep_b_frags<CIN, O><<<(NT * STEPS * 32 + 63) / 64, 64, 0, stream>>>(w2, bws);
    pac_conv_wmma<CIN, O, S, H><<<BATCH * Ho * (Ho / 64), 128, SMEM, stream>>>(ef1, bws, ef2);
    plane_stats<<<BATCH * O, 256, 0, stream>>>(ef2, stats, Ho * Ho);
    size_t tot = (size_t)BATCH * O * Ho * Ho;
    inorm_apply<<<(unsigned)((tot + 255) / 256), 256, 0, stream>>>(ef2, stats, 14, tot);
  }
  // ---------------- stage 3: pac_conv(ef2, ef2, w3, stride=2) ------------
  {
    constexpr int CIN = 64, O = 128, S = 2, H = 128, Ho = 64;
    constexpr int STEPS = (CIN * 9 + 31) / 32, NT = O / 16;
    constexpr int SMEM = CIN * 3 * (64 * S + 4) * 4;       // ~101.4 KB
    prep_b_frags<CIN, O><<<(NT * STEPS * 32 + 63) / 64, 64, 0, stream>>>(w3, bws);
    pac_conv_wmma<CIN, O, S, H><<<BATCH * Ho * (Ho / 64), 128, SMEM, stream>>>(ef2, bws, ef3);
    plane_stats<<<BATCH * O, 256, 0, stream>>>(ef3, stats, Ho * Ho);
    size_t tot = (size_t)BATCH * O * Ho * Ho;
    inorm_apply<<<(unsigned)((tot + 255) / 256), 256, 0, stream>>>(ef3, stats, 12, tot);
  }
}
